// CustomRNN_47674136986210
// MI455X (gfx1250) — matrix operand order repaired and checked by reading, not compiled, not run on hardware
//
#include <hip/hip_runtime.h>

// ---------------------------------------------------------------------------
// Stacked 2-layer tanh RNN for MI455X (gfx1250, wave32, WMMA).
//   B=64, T=2048, VOCAB=256, HID=512.
// Batch-sliced persistent workgroups (B/16 = 4 blocks x 1024 threads = 32
// waves; wave w owns output columns [16w,16w+16)).  Hidden states and the x
// tile live in LDS *pre-swizzled as bf16 hi/lo WMMA A-fragments*, so the hot
// loop is pure:  2x2 ds_load_b128 (A) + 2x2 global_load_b128 (B) + 3 WMMA.
// Chunk loops are kept rolled (unroll 2) so B-fragments stream from L2 through
// a small register window instead of being hoisted+spilled to scratch.
// ---------------------------------------------------------------------------

#define BATCH 64
#define TT    2048
#define NVOC  256
#define HID   512

typedef __attribute__((ext_vector_type(16))) __bf16 bhalf16;
typedef __attribute__((ext_vector_type(8)))  float  floatx8;

__device__ __forceinline__ floatx8 wmma_bf16(floatx8 acc, bhalf16 a, bhalf16 b) {
  return __builtin_amdgcn_wmma_f32_16x16x32_bf16(false, a, false, b,
                                                 (short)0, acc, false, false);
}

// A-fragment slot for K-position (within a 32-wide chunk) crel and row r:
//   lane = gc*16 + r,  element e = 2i+p   (16-bit 16x32 A-matrix ISA layout)
__device__ __forceinline__ int afrag_slot(int c, int crel, int r) {
  const int p  = crel & 1;
  const int gc = (crel >> 3) & 1;
  const int i  = ((crel & 16) >> 2) + ((crel & 7) >> 1);
  const int lane = gc * 16 + r;
  return (c * 32 + lane) * 16 + (2 * i + p);
}

__device__ __forceinline__ void split_bf16(float v, __bf16& hi, __bf16& lo) {
  hi = (__bf16)v;
  lo = (__bf16)(v - (float)hi);
}

// 16x16 tile accumulate: A fragments come pre-swizzled from LDS, B fragments
// stream from the packed global weight arrays.  K = 32*chunks.
// Loop kept rolled (unroll 2): per stage only A(16)+B(16)+acc(8) VGPRs live,
// so nothing spills and B stays an L2 stream.
__device__ __forceinline__ floatx8 gemm_tile(floatx8 acc,
                                             const __bf16* fhi, const __bf16* flo,
                                             const __bf16* __restrict__ ph,
                                             const __bf16* __restrict__ pl,
                                             int chunks, int tile, int lane) {
  const __bf16* bph = ph + ((size_t)tile * chunks * 32 + lane) * 16;
  const __bf16* bpl = pl + ((size_t)tile * chunks * 32 + lane) * 16;
  int fo = lane * 16;
#pragma unroll 2
  for (int c = 0; c < chunks; ++c) {
    const bhalf16 ah = *(const bhalf16*)(fhi + fo);     // 2x ds_load_b128
    const bhalf16 al = *(const bhalf16*)(flo + fo);
    const bhalf16 bh = *(const bhalf16*)bph;            // 2x global_load_b128
    const bhalf16 bl = *(const bhalf16*)bpl;
    acc = wmma_bf16(acc, ah, bh);   // hi*hi
    acc = wmma_bf16(acc, al, bh);   // lo*hi
    acc = wmma_bf16(acc, ah, bl);   // hi*lo  (lo*lo below f32 ulp)
    fo  += 32 * 16;                 // next chunk's fragment
    bph += 32 * 16;
    bpl += 32 * 16;
  }
  return acc;
}

// Pre-pack W (HID x kdim row-major; z = a @ W^T so B[k][n] = W[n][k]) into the
// wave32 WMMA B-fragment layout: lane l -> col n = tile*16 + l%16,
// K = 32c + 16*(j/8) + 8*(l/16) + j%8  (j = element within the lane's 16 bf16).
__global__ void pack_weight(const float* __restrict__ W, __bf16* __restrict__ ph,
                            __bf16* __restrict__ pl, int kdim, int total) {
  const int e = blockIdx.x * blockDim.x + threadIdx.x;
  if (e >= total) return;
  const int chunks = kdim >> 5;
  const int j    = e & 15;
  const int lane = (e >> 4) & 31;
  const int tc   = e >> 9;
  const int c    = tc % chunks;
  const int tile = tc / chunks;
  const int n = tile * 16 + (lane & 15);
  const int k = 32 * c + ((j & 8) << 1) + 8 * (lane >> 4) + (j & 7);
  const float w = W[(size_t)n * kdim + k];
  __bf16 hi, lo;
  split_bf16(w, hi, lo);
  ph[e] = hi;
  pl[e] = lo;
}

__global__ void copy_c(const float* __restrict__ c, float* __restrict__ dst, int n) {
  const int i = blockIdx.x * blockDim.x + threadIdx.x;
  if (i < n) dst[i] = c[i];
}

__global__ void __launch_bounds__(1024) rnn_persistent(
    const float* __restrict__ x,    // (B,T,NVOC)
    const float* __restrict__ hin,  // (2,B,HID)
    const float* __restrict__ b0, const float* __restrict__ b1,
    const __bf16* __restrict__ pWih0h, const __bf16* __restrict__ pWih0l,
    const __bf16* __restrict__ pWhh0h, const __bf16* __restrict__ pWhh0l,
    const __bf16* __restrict__ pWih1h, const __bf16* __restrict__ pWih1l,
    const __bf16* __restrict__ pWhh1h, const __bf16* __restrict__ pWhh1l,
    float* __restrict__ out) {
  // LDS: all activations stored as pre-swizzled bf16 hi/lo A-fragments.
  extern __shared__ __bf16 smem[];
  __bf16* xfh  = smem;              //  8 chunks * 512 = 4096
  __bf16* xfl  = xfh + 4096;
  __bf16* h0fh = xfl + 4096;        // 16 chunks * 512 = 8192
  __bf16* h0fl = h0fh + 8192;
  __bf16* h1fh = h0fl + 8192;
  __bf16* h1fl = h1fh + 8192;       // total 40960 bf16 = 80 KB

  const int tid   = threadIdx.x;
  const int lane  = tid & 31;
  const int wave  = tid >> 5;        // one 16-column output tile per wave
  const int m     = lane & 15;
  const int g     = lane >> 4;
  const int ncol  = wave * 16 + m;   // all 8 C elements of a lane share col ncol
  const int bbase = blockIdx.x * 16; // batch slice

  // h-fragment slot constants for this lane's output column (row varies by j)
  const int hc    = ncol >> 5;       // chunk
  const int hcrel = ncol & 31;

  // initial hidden states -> fragments
  for (int i = tid; i < 16 * HID; i += 1024) {
    const int r = i >> 9, n = i & (HID - 1);
    const int s = afrag_slot(n >> 5, n & 31, r);
    __bf16 hi, lo;
    split_bf16(hin[(size_t)(bbase + r) * HID + n], hi, lo);
    h0fh[s] = hi; h0fl[s] = lo;
    split_bf16(hin[(size_t)(BATCH + bbase + r) * HID + n], hi, lo);
    h1fh[s] = hi; h1fl[s] = lo;
  }
  const float bias0 = b0[ncol];
  const float bias1 = b1[ncol];
  __syncthreads();

  float* hout = out + (size_t)BATCH * TT * HID;  // (2,B,HID)

  for (int t = 0; t < TT; ++t) {
    // stage x[:, t, :] directly into bf16 hi/lo A-fragments (one cvt/element)
    for (int i = tid; i < 16 * NVOC; i += 1024) {
      const int r = i >> 8, k = i & (NVOC - 1);
      const int s = afrag_slot(k >> 5, k & 31, r);
      __bf16 hi, lo;
      split_bf16(x[((size_t)(bbase + r) * TT + t) * NVOC + k], hi, lo);
      xfh[s] = hi; xfl[s] = lo;
    }
    __syncthreads();  // x ready; prev-step h1 fragment writes visible

    // ---- layer 0: h0 = tanh(x @ Wih0^T + h0 @ Whh0^T + b0) ----
    floatx8 acc0 = {bias0, bias0, bias0, bias0, bias0, bias0, bias0, bias0};
    acc0 = gemm_tile(acc0, xfh,  xfl,  pWih0h, pWih0l, NVOC / 32, wave, lane);
    acc0 = gemm_tile(acc0, h0fh, h0fl, pWhh0h, pWhh0l, HID / 32,  wave, lane);
    float hn0[8];
#pragma unroll
    for (int j = 0; j < 8; ++j) hn0[j] = tanhf(acc0[j]);
    __syncthreads();  // all waves done reading h0 fragments
#pragma unroll
    for (int j = 0; j < 8; ++j) {   // scatter new h0 into its fragment slots
      const int s = afrag_slot(hc, hcrel, j + 8 * g);
      __bf16 hi, lo;
      split_bf16(hn0[j], hi, lo);
      h0fh[s] = hi; h0fl[s] = lo;
    }
    __syncthreads();  // new h0 fragments visible

    // ---- layer 1: h1 = tanh(h0 @ Wih1^T + h1 @ Whh1^T + b1) ----
    floatx8 acc1 = {bias1, bias1, bias1, bias1, bias1, bias1, bias1, bias1};
    acc1 = gemm_tile(acc1, h0fh, h0fl, pWih1h, pWih1l, HID / 32, wave, lane);
    acc1 = gemm_tile(acc1, h1fh, h1fl, pWhh1h, pWhh1l, HID / 32, wave, lane);
    float hn1[8];
#pragma unroll
    for (int j = 0; j < 8; ++j) hn1[j] = tanhf(acc1[j]);
    __syncthreads();  // all waves done reading h1 fragments
#pragma unroll
    for (int j = 0; j < 8; ++j) {
      const int row = j + 8 * g;
      const int s = afrag_slot(hc, hcrel, row);
      __bf16 hi, lo;
      split_bf16(hn1[j], hi, lo);
      h1fh[s] = hi; h1fl[s] = lo;
      out[((size_t)(bbase + row) * TT + t) * HID + ncol] = hn1[j];  // outs
    }
    if (t == TT - 1) {  // uniform branch: final states straight from registers
#pragma unroll
      for (int j = 0; j < 8; ++j) {
        const int row = j + 8 * g;
        hout[(size_t)(bbase + row) * HID + ncol]         = hn0[j];
        hout[(size_t)(BATCH + bbase + row) * HID + ncol] = hn1[j];
      }
    }
    // top-of-loop barrier orders these writes before next step's reads
  }
}

extern "C" void kernel_launch(void* const* d_in, const int* in_sizes, int n_in,
                              void* d_out, int out_size, void* d_ws, size_t ws_size,
                              hipStream_t stream) {
  const float* x    = (const float*)d_in[0];
  const float* h    = (const float*)d_in[1];
  const float* c    = (const float*)d_in[2];
  const float* Wih0 = (const float*)d_in[3];
  const float* Whh0 = (const float*)d_in[4];
  const float* b0   = (const float*)d_in[5];
  const float* Wih1 = (const float*)d_in[6];
  const float* Whh1 = (const float*)d_in[7];
  const float* b1   = (const float*)d_in[8];
  float* out = (float*)d_out;

  // workspace: packed bf16 hi/lo weight fragments (3.5 MB total)
  __bf16* ws = (__bf16*)d_ws;
  __bf16* pWih0h = ws;                   // 512*256
  __bf16* pWih0l = pWih0h + 131072;
  __bf16* pWhh0h = pWih0l + 131072;      // 512*512
  __bf16* pWhh0l = pWhh0h + 262144;
  __bf16* pWih1h = pWhh0l + 262144;
  __bf16* pWih1l = pWih1h + 262144;
  __bf16* pWhh1h = pWih1l + 262144;
  __bf16* pWhh1l = pWhh1h + 262144;

  pack_weight<<<131072 / 256, 256, 0, stream>>>(Wih0, pWih0h, pWih0l, NVOC, 131072);
  pack_weight<<<262144 / 256, 256, 0, stream>>>(Whh0, pWhh0h, pWhh0l, HID,  262144);
  pack_weight<<<262144 / 256, 256, 0, stream>>>(Wih1, pWih1h, pWih1l, HID,  262144);
  pack_weight<<<262144 / 256, 256, 0, stream>>>(Whh1, pWhh1h, pWhh1l, HID,  262144);

  const size_t lds_bytes = 40960 * sizeof(__bf16);  // 80 KB
  (void)hipFuncSetAttribute((const void*)rnn_persistent,
                            hipFuncAttributeMaxDynamicSharedMemorySize,
                            (int)lds_bytes);
  rnn_persistent<<<BATCH / 16, 1024, lds_bytes, stream>>>(
      x, h, b0, b1,
      pWih0h, pWih0l, pWhh0h, pWhh0l,
      pWih1h, pWih1l, pWhh1h, pWhh1l, out);

  // c pass-through, after outs (B*T*HID) and h_out (2*B*HID)
  float* cdst = out + (size_t)BATCH * TT * HID + (size_t)2 * BATCH * HID;
  copy_c<<<(2 * BATCH * HID + 255) / 256, 256, 0, stream>>>(c, cdst, 2 * BATCH * HID);
}
